// BidirectionalCrossAttention_70695161692374
// MI455X (gfx1250) — compile-verified, matrix operand order checked by reading, not run on hardware
//
#include <hip/hip_runtime.h>
#include <hip/hip_bf16.h>

#define BB 4
#define NN 1024
#define DD 1024
#define HH 16
#define DHD 64
#define INNER 1024

typedef __attribute__((ext_vector_type(16))) __bf16 v16bf;
typedef __attribute__((ext_vector_type(8)))  float  v8f;

union Frag { v16bf v; unsigned int u[8]; };
union U4   { uint4 v; unsigned short h[8]; };

__device__ inline unsigned short f2bf(float f) {
  unsigned int u = __builtin_bit_cast(unsigned int, f);
  u += 0x7FFFu + ((u >> 16) & 1u);      // round-to-nearest-even
  return (unsigned short)(u >> 16);
}

// CDNA5 async copy: global -> LDS, 16B per lane, tracked by ASYNCcnt.
__device__ inline void async_b128(const void* g, const void* l) {
  unsigned lds = (unsigned)(unsigned long long)l;   // flat shared addr[31:0] = LDS offset
  asm volatile("global_load_async_to_lds_b128 %0, %1, off"
               :: "v"(lds), "v"(g) : "memory");
}
__device__ inline void wait_async0() {
#if __has_builtin(__builtin_amdgcn_s_wait_asynccnt)
  __builtin_amdgcn_s_wait_asynccnt(0);
#else
  asm volatile("s_wait_asynccnt 0" ::: "memory");
#endif
}

// Load a 16x32 bf16 WMMA operand fragment from LDS.
// base: row-major [16][ld] slab (A: m x k). B operands are staged [n][k],
// so the same loader works. Per ISA 7.12.2 16-bit A layout.
__device__ inline v16bf ld_frag(const unsigned short* base, int ld, int lane) {
  const int m = lane & 15, half = lane >> 4;
  Frag f;
  const unsigned short* row = base + m * ld + half * 8;
#pragma unroll
  for (int r = 0; r < 4; ++r) {
    f.u[r]     = *(const unsigned int*)(row + 2 * r);
    f.u[r + 4] = *(const unsigned int*)(row + 16 + 2 * r);
  }
  return f.v;
}

__device__ inline v8f wmma_bf16(v16bf a, v16bf b, v8f c) {
  return __builtin_amdgcn_wmma_f32_16x16x32_bf16(false, a, false, b, (short)0, c, false, false);
}

// ---------------------------------------------------------------- convert
__global__ void k_convert(const float* __restrict__ in,
                          unsigned short* __restrict__ out, int n4) {
  int i = blockIdx.x * blockDim.x + threadIdx.x;
  if (i < n4) {
    float4 f = ((const float4*)in)[i];
    union { unsigned short h[4]; unsigned long long u; } r;
    r.h[0] = f2bf(f.x); r.h[1] = f2bf(f.y); r.h[2] = f2bf(f.z); r.h[3] = f2bf(f.w);
    ((unsigned long long*)out)[i] = r.u;
  }
}

// ---------------------------------------------------------------- GEMM
// C[M,Nc] = A0*B0 (+ A1*B1) (+ bias0 + bias1), bf16 in, bf16 or f32 out.
// 128x128 C tile / 128 threads (4 waves, each a 64x64 slab = 4x4 wmma tiles:
// every A/B fragment is reused 4x from registers -> 2 WMMA per fragment load).
// Double-buffered LDS; A tile prefetched via async-to-LDS, B tile via wide
// global loads held in regs through the compute phase, then scattered [n][k].
__global__ __launch_bounds__(128) void k_gemm(
    const unsigned short* __restrict__ A0, const unsigned short* __restrict__ B0,
    const unsigned short* __restrict__ A1, const unsigned short* __restrict__ B1,
    const float* __restrict__ bias0, const float* __restrict__ bias1,
    unsigned short* __restrict__ outb, float* __restrict__ outf,
    int M, int Nc, int K) {
  __shared__ unsigned short As[2][128 * 32];   // [m][k]
  __shared__ unsigned short Bt[2][128 * 32];   // [n][k]
  const int t = threadIdx.x, lane = t & 31, wid = t >> 5;
  const int wy = wid & 1, wx = wid >> 1;       // 2x2 waves of 64x64
  const int lane_n = lane & 15, half = lane >> 4;
  const int m0 = blockIdx.y * 128, n0 = blockIdx.x * 128;
  const int kpp = K / 32;
  const int total = (A1 ? 2 : 1) * kpp;
  v8f acc[4][4] = {};
  (void)M;

  auto srcAB = [&](int s, const unsigned short*& A, const unsigned short*& B, int& k0) {
    A = (s >= kpp) ? A1 : A0;
    B = (s >= kpp) ? B1 : B0;
    k0 = ((s >= kpp) ? s - kpp : s) * 32;
  };

  {  // prologue: stage step 0 into buffer 0
    const unsigned short *A, *B; int k0;
    srcAB(0, A, B, k0);
#pragma unroll
    for (int i = 0; i < 4; ++i) {              // A: 512 b128 chunks / 128 thr
      int lin = t + i * 128;
      int r = lin >> 2, seg = (lin & 3) * 8;
      async_b128(A + (size_t)(m0 + r) * K + k0 + seg, &As[0][lin * 8]);
    }
#pragma unroll
    for (int i = 0; i < 4; ++i) {              // B: wide load + transpose scatter
      int lin = t + i * 128;
      int kr = lin >> 4, nc8 = (lin & 15) * 8;
      U4 d; d.v = *(const uint4*)(B + (size_t)(k0 + kr) * Nc + n0 + nc8);
#pragma unroll
      for (int e = 0; e < 8; ++e) Bt[0][(nc8 + e) * 32 + kr] = d.h[e];
    }
  }
  wait_async0();
  __syncthreads();

  for (int s = 0; s < total; ++s) {
    const int cur = s & 1, nxt = cur ^ 1;
    const bool pre = (s + 1 < total);
    U4 breg[4]; int bkr[4], bnc8[4];
    if (pre) {                                 // issue prefetch for s+1
      const unsigned short *A, *B; int k0;
      srcAB(s + 1, A, B, k0);
#pragma unroll
      for (int i = 0; i < 4; ++i) {
        int lin = t + i * 128;
        int r = lin >> 2, seg = (lin & 3) * 8;
        async_b128(A + (size_t)(m0 + r) * K + k0 + seg, &As[nxt][lin * 8]);
      }
#pragma unroll
      for (int i = 0; i < 4; ++i) {
        int lin = t + i * 128;
        bkr[i] = lin >> 4; bnc8[i] = (lin & 15) * 8;
        breg[i].v = *(const uint4*)(B + (size_t)(k0 + bkr[i]) * Nc + n0 + bnc8[i]);
      }
    }
    // compute on buffer cur: 8 fragment loads feed 16 WMMAs
    v16bf bf[4];
#pragma unroll
    for (int nt = 0; nt < 4; ++nt)
      bf[nt] = ld_frag(&Bt[cur][(wx * 64 + nt * 16) * 32], 32, lane);
#pragma unroll
    for (int mt = 0; mt < 4; ++mt) {
      v16bf af = ld_frag(&As[cur][(wy * 64 + mt * 16) * 32], 32, lane);
#pragma unroll
      for (int nt = 0; nt < 4; ++nt)
        acc[mt][nt] = wmma_bf16(af, bf[nt], acc[mt][nt]);
    }
    if (pre) {                                 // drain prefetch into LDS
#pragma unroll
      for (int i = 0; i < 4; ++i)
#pragma unroll
        for (int e = 0; e < 8; ++e)
          Bt[nxt][(bnc8[i] + e) * 32 + bkr[i]] = breg[i].h[e];
      wait_async0();
    }
    __syncthreads();
  }

  // epilogue: C/D layout = lane holds col (lane&15), rows r + 8*(lane>>4)
#pragma unroll
  for (int mt = 0; mt < 4; ++mt) {
#pragma unroll
    for (int nt = 0; nt < 4; ++nt) {
      int col = n0 + wx * 64 + nt * 16 + lane_n;
      float badd = 0.f;
      if (bias0) badd += bias0[col];
      if (bias1) badd += bias1[col];
#pragma unroll
      for (int r = 0; r < 8; ++r) {
        int row = m0 + wy * 64 + mt * 16 + half * 8 + r;
        float val = acc[mt][nt][r] + badd;
        if (outf) outf[(size_t)row * Nc + col] = val;
        else      outb[(size_t)row * Nc + col] = f2bf(val);
      }
    }
  }
}

// ---------------------------------------------------------------- attention
// One block per (b, h, 64-row i-tile). Single pass over 16 j-tiles computing
// E = exp(scale*sim) as row-block E[i,:] and col-block E[:,i], accumulating
// out_acc = E*cv, ctx_acc = E^T*v, rowsum, colsum. Normalize at the end.
__global__ __launch_bounds__(256) void k_attn(
    const unsigned short* __restrict__ qk, const unsigned short* __restrict__ cqk,
    const unsigned short* __restrict__ vv, const unsigned short* __restrict__ cv,
    unsigned short* __restrict__ outm, unsigned short* __restrict__ ctxm) {
  __shared__ unsigned short Qi[64 * 64];   // qk[i-tile]  [i][d]
  __shared__ unsigned short Ci[64 * 64];   // cqk[i-tile] [i][d]
  __shared__ unsigned short TA[64 * 64];   // stage1: Qj [j][d]; stage2: Vt [d][j]
  __shared__ unsigned short TB[64 * 64];   // stage1: Cj [j][d]; stage2: CVt [d][j]
  __shared__ unsigned short E1s[64 * 64];  // exp(sim[i,j])  [i][j]
  __shared__ unsigned short E2s[64 * 64];  // exp(sim[j,i])  stored transposed [i][j]
  __shared__ float rowsum[64], colsum[64];

  const int t = threadIdx.x, lane = t & 31, wid = t >> 5;
  const int lane_n = lane & 15, half = lane >> 4;
  const int i0 = blockIdx.x * 64, h = blockIdx.y, b = blockIdx.z;
  const size_t rowbase = (size_t)(b * NN) * INNER + (size_t)h * DHD;

#pragma unroll
  for (int i = 0; i < 2; ++i) {              // Qi, Ci via async b128 chunks
    int lin = t + i * 256;
    int r = lin >> 3, seg = (lin & 7) * 8;
    size_t g = rowbase + (size_t)(i0 + r) * INNER + seg;
    async_b128(qk + g,  &Qi[lin * 8]);
    async_b128(cqk + g, &Ci[lin * 8]);
  }
  if (t < 64) { rowsum[t] = 0.f; colsum[t] = 0.f; }
  v8f oacc[4] = {};   // wid<4: out_acc rows; wid>=4: ctx_acc rows
  wait_async0();
  __syncthreads();

  for (int jt = 0; jt < 16; ++jt) {
    const int j0 = jt * 64;
#pragma unroll
    for (int i = 0; i < 2; ++i) {            // Qj -> TA, Cj -> TB (async)
      int lin = t + i * 256;
      int r = lin >> 3, seg = (lin & 7) * 8;
      size_t g = rowbase + (size_t)(j0 + r) * INNER + seg;
      async_b128(qk + g,  &TA[lin * 8]);
      async_b128(cqk + g, &TB[lin * 8]);
    }
    wait_async0();
    __syncthreads();

    if (wid < 4) {   // S1[i,j] = Qi . Cj^T : 16-row slab, 64 cols, K=64
      v8f s[4] = {};
#pragma unroll
      for (int ks = 0; ks < 2; ++ks) {
        v16bf a = ld_frag(Qi + (wid * 16) * 64 + ks * 32, 64, lane);
#pragma unroll
        for (int nt = 0; nt < 4; ++nt) {
          v16bf bb = ld_frag(TB + (nt * 16) * 64 + ks * 32, 64, lane);
          s[nt] = wmma_bf16(a, bb, s[nt]);
        }
      }
      float rs[8] = {};
#pragma unroll
      for (int nt = 0; nt < 4; ++nt)
#pragma unroll
        for (int r = 0; r < 8; ++r) {
          float e = __expf(s[nt][r] * 0.125f);   // scale = DH^-0.5 = 1/8
          int row = wid * 16 + half * 8 + r;
          E1s[row * 64 + nt * 16 + lane_n] = f2bf(e);
          rs[r] += e;
        }
#pragma unroll
      for (int r = 0; r < 8; ++r)
        atomicAdd(&rowsum[wid * 16 + half * 8 + r], rs[r]);
    } else {         // S2[j,i] = Qj . Ci^T : store transposed, colsum over j
      v8f s[4] = {};
#pragma unroll
      for (int ks = 0; ks < 2; ++ks) {
        v16bf a = ld_frag(TA + ((wid - 4) * 16) * 64 + ks * 32, 64, lane);
#pragma unroll
        for (int nt = 0; nt < 4; ++nt) {
          v16bf bb = ld_frag(Ci + (nt * 16) * 64 + ks * 32, 64, lane);
          s[nt] = wmma_bf16(a, bb, s[nt]);
        }
      }
#pragma unroll
      for (int nt = 0; nt < 4; ++nt) {
        float cs = 0.f;
#pragma unroll
        for (int r = 0; r < 8; ++r) {
          float e = __expf(s[nt][r] * 0.125f);
          int jrow = (wid - 4) * 16 + half * 8 + r;
          E2s[(nt * 16 + lane_n) * 64 + jrow] = f2bf(e);
          cs += e;
        }
        atomicAdd(&colsum[nt * 16 + lane_n], cs);
      }
    }
    __syncthreads();

#pragma unroll
    for (int i = 0; i < 2; ++i) {            // Vt -> TA, CVt -> TB ([d][j])
      int lin = t + i * 256;
      int r = lin >> 3, c8 = (lin & 7) * 8;  // r = j, c8 = d base
      size_t g = rowbase + (size_t)(j0 + r) * INNER + c8;
      U4 dv, dc;
      dv.v = *(const uint4*)(vv + g);
      dc.v = *(const uint4*)(cv + g);
#pragma unroll
      for (int e = 0; e < 8; ++e) {
        TA[(c8 + e) * 64 + r] = dv.h[e];
        TB[(c8 + e) * 64 + r] = dc.h[e];
      }
    }
    __syncthreads();

    if (wid < 4) {   // out_acc += E1 * cv
#pragma unroll
      for (int ks = 0; ks < 2; ++ks) {
        v16bf a = ld_frag(E1s + (wid * 16) * 64 + ks * 32, 64, lane);
#pragma unroll
        for (int nt = 0; nt < 4; ++nt) {
          v16bf bb = ld_frag(TB + (nt * 16) * 64 + ks * 32, 64, lane);
          oacc[nt] = wmma_bf16(a, bb, oacc[nt]);
        }
      }
    } else {         // ctx_acc += E2^T * v
#pragma unroll
      for (int ks = 0; ks < 2; ++ks) {
        v16bf a = ld_frag(E2s + ((wid - 4) * 16) * 64 + ks * 32, 64, lane);
#pragma unroll
        for (int nt = 0; nt < 4; ++nt) {
          v16bf bb = ld_frag(TA + (nt * 16) * 64 + ks * 32, 64, lane);
          oacc[nt] = wmma_bf16(a, bb, oacc[nt]);
        }
      }
    }
    __syncthreads();
  }

  if (wid < 4) {
#pragma unroll
    for (int nt = 0; nt < 4; ++nt)
#pragma unroll
      for (int r = 0; r < 8; ++r) {
        int il = wid * 16 + half * 8 + r;
        float val = oacc[nt][r] / rowsum[il];
        outm[rowbase + (size_t)(i0 + il) * INNER + nt * 16 + lane_n] = f2bf(val);
      }
  } else {
#pragma unroll
    for (int nt = 0; nt < 4; ++nt)
#pragma unroll
      for (int r = 0; r < 8; ++r) {
        int il = (wid - 4) * 16 + half * 8 + r;
        float val = oacc[nt][r] / colsum[il];
        ctxm[rowbase + (size_t)(i0 + il) * INNER + nt * 16 + lane_n] = f2bf(val);
      }
  }
}

// ---------------------------------------------------------------- LN + residual
__global__ __launch_bounds__(256) void k_ln(const float* __restrict__ y,
    const float* __restrict__ x, const float* __restrict__ ctx,
    const float* __restrict__ gamma, const float* __restrict__ beta,
    float* __restrict__ out) {
  __shared__ float sA[256], sB[256];
  const int t = threadIdx.x;
  const size_t row = blockIdx.x;
  float4 v = ((const float4*)(y + row * DD))[t];
  sA[t] = v.x + v.y + v.z + v.w;
  sB[t] = v.x * v.x + v.y * v.y + v.z * v.z + v.w * v.w;
  __syncthreads();
  for (int off = 128; off > 0; off >>= 1) {
    if (t < off) { sA[t] += sA[t + off]; sB[t] += sB[t + off]; }
    __syncthreads();
  }
  const float mean = sA[0] * (1.f / (float)DD);
  const float var  = sB[0] * (1.f / (float)DD) - mean * mean;
  const float rstd = rsqrtf(var + 1e-5f);
  float4 g = ((const float4*)gamma)[t];
  float4 be = ((const float4*)beta)[t];
  float4 xr = ((const float4*)(x + row * DD))[t];
  float4 cr = ((const float4*)(ctx + row * DD))[t];
  float4 o;
  o.x = g.x * (v.x - mean) * rstd + be.x + xr.x + cr.x;
  o.y = g.y * (v.y - mean) * rstd + be.y + xr.y + cr.y;
  o.z = g.z * (v.z - mean) * rstd + be.z + xr.z + cr.z;
  o.w = g.w * (v.w - mean) * rstd + be.w + xr.w + cr.w;
  ((float4*)(out + row * DD))[t] = o;
}

// ---------------------------------------------------------------- launch
extern "C" void kernel_launch(void* const* d_in, const int* in_sizes, int n_in,
                              void* d_out, int out_size, void* d_ws, size_t ws_size,
                              hipStream_t stream) {
  (void)in_sizes; (void)n_in; (void)out_size; (void)ws_size;
  const float* x     = (const float*)d_in[0];
  const float* ctx   = (const float*)d_in[1];
  const float* Wqk   = (const float*)d_in[2];
  const float* Wcqk  = (const float*)d_in[3];
  const float* Wv    = (const float*)d_in[4];
  const float* Wcv   = (const float*)d_in[5];
  const float* Wout  = (const float*)d_in[6];
  const float* bout  = (const float*)d_in[7];
  const float* Wcout = (const float*)d_in[8];
  const float* bcout = (const float*)d_in[9];
  const float* gamma = (const float*)d_in[10];
  const float* beta  = (const float*)d_in[11];
  float* out = (float*)d_out;

  const size_t SXN = (size_t)BB * NN * DD;  // 4M elements
  const size_t SW  = (size_t)DD * INNER;    // 1M elements
  unsigned short* w = (unsigned short*)d_ws;
  unsigned short* xb    = w; w += SXN;
  unsigned short* cb    = w; w += SXN;
  unsigned short* qkb   = w; w += SXN;
  unsigned short* vb    = w; w += SXN;
  unsigned short* cqkb  = w; w += SXN;
  unsigned short* cvb   = w; w += SXN;
  unsigned short* outm  = w; w += SXN;
  unsigned short* ctxm  = w; w += SXN;
  unsigned short* wqk   = w; w += SW;
  unsigned short* wcqk  = w; w += SW;
  unsigned short* wv    = w; w += SW;
  unsigned short* wcv   = w; w += SW;
  unsigned short* wout  = w; w += SW;
  unsigned short* wcout = w; w += SW;
  float* yproj = (float*)w;                 // 4M floats (16 MB)

  auto cvt = [&](const float* src, unsigned short* dst, size_t n) {
    int n4 = (int)(n / 4);
    k_convert<<<(n4 + 255) / 256, 256, 0, stream>>>(src, dst, n4);
  };
  cvt(x, xb, SXN);       cvt(ctx, cb, SXN);
  cvt(Wqk, wqk, SW);     cvt(Wcqk, wcqk, SW);
  cvt(Wv, wv, SW);       cvt(Wcv, wcv, SW);
  cvt(Wout, wout, SW);   cvt(Wcout, wcout, SW);

  dim3 gg(INNER / 128, (BB * NN) / 128);    // (8, 32)
  k_gemm<<<gg, 128, 0, stream>>>(xb, wqk,  nullptr, nullptr, nullptr, nullptr, qkb,  nullptr, BB*NN, INNER, DD);
  k_gemm<<<gg, 128, 0, stream>>>(xb, wv,   nullptr, nullptr, nullptr, nullptr, vb,   nullptr, BB*NN, INNER, DD);
  k_gemm<<<gg, 128, 0, stream>>>(cb, wcqk, nullptr, nullptr, nullptr, nullptr, cqkb, nullptr, BB*NN, INNER, DD);
  k_gemm<<<gg, 128, 0, stream>>>(cb, wcv,  nullptr, nullptr, nullptr, nullptr, cvb,  nullptr, BB*NN, INNER, DD);

  k_attn<<<dim3(NN / 64, HH, BB), 256, 0, stream>>>(qkb, cqkb, vb, cvb, outm, ctxm);

  k_gemm<<<dim3(DD / 128, (BB * NN) / 128), 128, 0, stream>>>(
      outm, wout, ctxm, wcout, bout, bcout, nullptr, yproj, BB*NN, DD, INNER);

  k_ln<<<BB * NN, 256, 0, stream>>>(yproj, x, ctx, gamma, beta, out);
}
